// ButterflyLayer_30296699306355
// MI455X (gfx1250) — matrix operand order repaired and checked by reading, not compiled, not run on hardware
//
#include <hip/hip_runtime.h>
#include <hip/hip_bf16.h>

// ---------------------------------------------------------------------------
// Butterfly net on gfx1250 (MI455X), CDNA5 WMMA path.
//
// Activations: flat bf16 [ROWS=262144][C=64], rows ordered (branch,batch,time).
// Per level, parent branch k gives A_k[M_l x 128] (two consecutive parent rows
// = 128 contiguous bf16) x W[kc][128 x 64] -> contiguous child block.
//
// Level kernel: grid = 1024 blocks x 256 threads (8 waves). Block = one
// (child kc, 256-row M tile). Each wave owns 32 rows x 64 cols:
//   4 K-steps x { 2 A frags (global b128), 4 B frags (ds b128),
//                 8 x v_wmma_f32_16x16x32_bf16 }  = 32 WMMA / wave,
// f32 accumulation, bias+relu, bf16 store.
//
// d_ws requirement: 2 * 262144 * 64 * 2 bytes = 64 MiB ping-pong.
// ---------------------------------------------------------------------------

typedef __attribute__((ext_vector_type(16))) __bf16 v16bf;
typedef __attribute__((ext_vector_type(8)))  __bf16 v8bf;
typedef __attribute__((ext_vector_type(8)))  float  v8f;

#define NLVL        10
#define CHN         64
#define ROWS        262144        // K * B * T at every level boundary
#define LDS_STRIDE  136           // 128 K-cols + 8 pad (bank spreading)

// ---- input layer: [B,2048,1] --(pairs @ w_in[2x64], relu)--> bf16 acts -----
__global__ __launch_bounds__(256) void bfly_input_kernel(
    const float* __restrict__ in,     // [256][2048]
    const float* __restrict__ w_in,   // [2][1][64]
    const float* __restrict__ b_in,   // [64]
    __bf16* __restrict__ act)         // [ROWS][64], rows = b*1024 + t (k=0)
{
  int gid = blockIdx.x * blockDim.x + threadIdx.x;   // 16,777,216 total
  int c   = gid & 63;
  int row = gid >> 6;                                // b*1024 + t
  int b   = row >> 10;
  int t   = row & 1023;
  const float* p = in + b * 2048 + 2 * t;
  float v = p[0] * w_in[c] + p[1] * w_in[64 + c] + b_in[c];
  v = fmaxf(v, 0.0f);
  act[(size_t)row * CHN + c] = (__bf16)v;
}

// ---- one butterfly level: K parent GEMMs -> 2K child blocks ----------------
__global__ __launch_bounds__(256) void bfly_level_kernel(
    const __bf16* __restrict__ act_in,   // parent acts [ROWS][64]
    __bf16*       __restrict__ act_out,  // child acts  [ROWS][64]
    const float*  __restrict__ W,        // filters[l]: [2K][2][64][64] f32
    const float*  __restrict__ bias,     // biases[l]:  [2K][64] f32
    int lvl)
{
  __shared__ __bf16 wt[CHN * LDS_STRIDE];            // W[kc]^T as [N=64][K=128]

  const int kc_shift = (NLVL - 1) - lvl;             // tiles_per_child = 512>>lvl
  const int tpc   = 1 << kc_shift;
  const int kc    = blockIdx.x >> kc_shift;          // child index (= 2k+c)
  const int mtile = blockIdx.x & (tpc - 1);          // 256-row M tile
  const int k     = kc >> 1;                         // parent branch
  const int Ml    = 131072 >> lvl;                   // pair-rows per parent

  // ---- stage weight block (32 KB f32) into LDS, transposed + bf16 ----------
  const float* wblk = W + (size_t)kc * 8192;         // [2][64][64]
  #pragma unroll
  for (int j = 0; j < 32; ++j) {
    int idx = threadIdx.x + j * 256;                 // 0..8191
    int p   = idx >> 12;                             // tap
    int rem = idx & 4095;
    int i   = rem >> 6;                              // Cin
    int o   = rem & 63;                              // Cout
    wt[o * LDS_STRIDE + p * CHN + i] = (__bf16)wblk[idx];
  }
  __syncthreads();

  const int wave  = threadIdx.x >> 5;
  const int lane  = threadIdx.x & 31;
  const int mrow  = lane & 15;                       // M row / N col selector
  const int khalf = lane >> 4;

  // two 16-row M subtiles per wave: rows m0 and m0+16 within the branch
  const int m0 = mtile * 256 + wave * 32 + mrow;
  const __bf16* arow0 = act_in + (size_t)k * Ml * 128 + (size_t)m0 * 128;
  const __bf16* arow1 = arow0 + (size_t)16 * 128;

  v8f acc[2][4];
  #pragma unroll
  for (int u = 0; u < 2; ++u)
    #pragma unroll
    for (int n = 0; n < 4; ++n)
      acc[u][n] = (v8f){};

  #pragma unroll
  for (int s = 0; s < 4; ++s) {                      // K = 128 in 4 steps of 32
    const int ks = s * 32;
    // A fragments (16x32 bf16 each): lane holds K in [kh*8,+8) and +16
    union { v16bf v; v8bf h[2]; } a0, a1;
    a0.h[0] = *(const v8bf*)(arow0 + ks + khalf * 8);
    a0.h[1] = *(const v8bf*)(arow0 + ks + khalf * 8 + 16);
    a1.h[0] = *(const v8bf*)(arow1 + ks + khalf * 8);
    a1.h[1] = *(const v8bf*)(arow1 + ks + khalf * 8 + 16);
    // B fragments (32x16 bf16): lane holds col N, K in [kh*8,+8) and +16
    const int kb = ks + khalf * 8;
    union { v16bf v; v8bf h[2]; } bfr[4];
    #pragma unroll
    for (int n = 0; n < 4; ++n) {
      const __bf16* bp = &wt[(n * 16 + mrow) * LDS_STRIDE + kb];
      bfr[n].h[0] = *(const v8bf*)(bp);
      bfr[n].h[1] = *(const v8bf*)(bp + 16);
    }
    // 8 back-to-back WMMAs: B fragment reused across both M subtiles
    #pragma unroll
    for (int n = 0; n < 4; ++n) {
      acc[0][n] = __builtin_amdgcn_wmma_f32_16x16x32_bf16(
          false, a0.v, false, bfr[n].v, (short)0, acc[0][n], false, false);
      acc[1][n] = __builtin_amdgcn_wmma_f32_16x16x32_bf16(
          false, a1.v, false, bfr[n].v, (short)0, acc[1][n], false, false);
    }
  }

  // ---- epilogue: bias + relu + bf16 store (C/D layout: M = kh*8+r, N = mrow)
  const float* brow = bias + kc * CHN;
  const float  b0 = brow[ 0 + mrow];
  const float  b1 = brow[16 + mrow];
  const float  b2 = brow[32 + mrow];
  const float  b3 = brow[48 + mrow];
  __bf16* oblk = act_out + ((size_t)kc * Ml + (size_t)mtile * 256 + wave * 32) * CHN;
  #pragma unroll
  for (int u = 0; u < 2; ++u) {
    #pragma unroll
    for (int r = 0; r < 8; ++r) {
      const int M = u * 16 + khalf * 8 + r;
      __bf16* orow = oblk + (size_t)M * CHN;
      orow[ 0 + mrow] = (__bf16)fmaxf(acc[u][0][r] + b0, 0.0f);
      orow[16 + mrow] = (__bf16)fmaxf(acc[u][1][r] + b1, 0.0f);
      orow[32 + mrow] = (__bf16)fmaxf(acc[u][2][r] + b2, 0.0f);
      orow[48 + mrow] = (__bf16)fmaxf(acc[u][3][r] + b3, 0.0f);
    }
  }
}

// ---- dense heads: per branch [256x64] @ [64x2] -> out [B,2048] -------------
__global__ __launch_bounds__(256) void bfly_head_kernel(
    const __bf16* __restrict__ act,   // [ROWS][64], row = kc*256 + b
    const float*  __restrict__ fea,   // [1024][64][2]
    float*        __restrict__ out)   // [256][2048]
{
  int r  = blockIdx.x * blockDim.x + threadIdx.x;    // 0..262143
  int kc = r >> 8;
  int b  = r & 255;
  const __bf16* a = act + (size_t)r * CHN;
  const float*  w = fea + (size_t)kc * 128;
  float s0 = 0.0f, s1 = 0.0f;
  #pragma unroll
  for (int i = 0; i < CHN; ++i) {
    float av = (float)a[i];
    s0 += av * w[2 * i];
    s1 += av * w[2 * i + 1];
  }
  out[(size_t)b * 2048 + kc * 2 + 0] = s0;
  out[(size_t)b * 2048 + kc * 2 + 1] = s1;
}

// ---------------------------------------------------------------------------
extern "C" void kernel_launch(void* const* d_in, const int* in_sizes, int n_in,
                              void* d_out, int out_size, void* d_ws, size_t ws_size,
                              hipStream_t stream) {
  (void)in_sizes; (void)n_in; (void)out_size; (void)ws_size;
  // dict order: in_data, w_in, b_in, filters[0..9], biases[0..9], fea_dense
  const float* in_data = (const float*)d_in[0];
  const float* w_in    = (const float*)d_in[1];
  const float* b_in    = (const float*)d_in[2];
  const float* fea     = (const float*)d_in[3 + NLVL + NLVL];

  __bf16* acta = (__bf16*)d_ws;                         // 32 MiB
  __bf16* actb = acta + (size_t)ROWS * CHN;             // 32 MiB
  __bf16* bufs[2] = {acta, actb};

  // input projection -> bufs[0]
  bfly_input_kernel<<<(ROWS * CHN) / 256, 256, 0, stream>>>(in_data, w_in, b_in, bufs[0]);

  // 10 butterfly levels, ping-pong
  for (int l = 0; l < NLVL; ++l) {
    bfly_level_kernel<<<1024, 256, 0, stream>>>(
        bufs[l & 1], bufs[(l + 1) & 1],
        (const float*)d_in[3 + l],
        (const float*)d_in[3 + NLVL + l],
        l);
  }

  // heads read bufs[NLVL % 2] == bufs[0]
  bfly_head_kernel<<<ROWS / 256, 256, 0, stream>>>(bufs[0], fea, (float*)d_out);
}
